// MambaBlock2D_48172353192409
// MI455X (gfx1250) — matrix operand order repaired and checked by reading, not compiled, not run on hardware
//
#include <hip/hip_runtime.h>

// ---------------------------------------------------------------------------
// MambaBlock2D for gfx1250 (MI455X): bf16 WMMA GEMMs (async-LDS double-buffered)
// + VALU selective-scan pipeline.
// ---------------------------------------------------------------------------

typedef __attribute__((ext_vector_type(16))) __bf16 v16bf;
typedef __attribute__((ext_vector_type(8)))  float  v8f;

union FragBF { v16bf v; unsigned u[8]; };

constexpr int C_DIM = 256;          // DIM
constexpr int DI    = 512;          // D_INNER
constexpr int NST   = 16;           // D_STATE
constexpr int RNK   = 16;           // DT_RANK
constexpr int LTOK  = 4096;         // H*W
constexpr int PROJW = RNK + 2*NST;  // 48

// CDNA5 async global->LDS path (ASYNCcnt-tracked), if the toolchain exposes it.
#ifdef __has_builtin
#  if __has_builtin(__builtin_amdgcn_global_load_async_to_lds_b128)
#    define USE_ASYNC_LDS 1
#  endif
#endif
#ifndef USE_ASYNC_LDS
#  define USE_ASYNC_LDS 0
#endif

#if USE_ASYNC_LDS
// Parameter type per hipcc diagnostic: pointer to int __attribute__((vector_size(16)))
// in global address space; LDS side in address_space(3).
typedef int async_v4i __attribute__((vector_size(16)));
typedef __attribute__((address_space(1))) async_v4i* gptr_v4i;
typedef __attribute__((address_space(3))) async_v4i* lptr_v4i;
#endif

__device__ __forceinline__ void wait_asynccnt0() {
#if USE_ASYNC_LDS
  asm volatile("s_wait_asynccnt 0x0" ::: "memory");
#endif
}

// ---------------- helpers ----------------

__device__ __forceinline__ __bf16 f2bf(float f) {
  union { float f; unsigned u; } a; a.f = f;
  unsigned r = a.u + 0x7fffu + ((a.u >> 16) & 1u);   // round-to-nearest-even
  unsigned short h = (unsigned short)(r >> 16);
  union { unsigned short s; __bf16 b; } o; o.s = h;
  return o.b;
}

__device__ __forceinline__ float siluf(float x) { return x / (1.f + __expf(-x)); }
__device__ __forceinline__ float geluf(float x) { return 0.5f * x * (1.f + erff(x * 0.70710678118654752f)); }
__device__ __forceinline__ float softplusf(float x) { return (x > 20.f) ? x : log1pf(__expf(x)); }

// ---------------- f32 -> bf16 conversion ----------------

__global__ void cvt_bf16_kernel(const float* __restrict__ in, __bf16* __restrict__ out, int n) {
  int i = blockIdx.x * blockDim.x + threadIdx.x;
  if (i < n) out[i] = f2bf(in[i]);
}

// ---------------- LayerNorm (one token per block) ----------------

__global__ __launch_bounds__(256) void ln_kernel(
    const float* __restrict__ x, const float* __restrict__ gamma, const float* __restrict__ beta,
    float* __restrict__ ynF, __bf16* __restrict__ ynB) {
  const int token = blockIdx.x;               // b*LTOK + l
  const int b = token / LTOK;
  const int l = token - b * LTOK;
  const int c = threadIdx.x;                  // 0..255

  __shared__ float red[256];
  float v = x[((size_t)(b * C_DIM + c)) * LTOK + l];

  red[c] = v; __syncthreads();
  #pragma unroll
  for (int s = 128; s > 0; s >>= 1) { if (c < s) red[c] += red[c + s]; __syncthreads(); }
  float mu = red[0] * (1.f / C_DIM);
  __syncthreads();

  float dv = v - mu;
  red[c] = dv * dv; __syncthreads();
  #pragma unroll
  for (int s = 128; s > 0; s >>= 1) { if (c < s) red[c] += red[c + s]; __syncthreads(); }
  float var = red[0] * (1.f / C_DIM);

  float yn = dv * rsqrtf(var + 1e-5f) * gamma[c] + beta[c];
  size_t o = (size_t)token * C_DIM + c;
  ynF[o] = yn;
  ynB[o] = f2bf(yn);
}

// ---------------- WMMA GEMM: C[M,N] = A[M,K] * B[N,K]^T (+ epilogue) ----------------

#define GBM 64
#define GBN 128
#define GBK 32
#define LDT 48   // padded LDS row stride in bf16 elements (96B -> good bank spread, 16B aligned)

enum { EPI_STORE = 0, EPI_RESID = 1, EPI_BIAS_GELU = 2, EPI_BIAS_RESID_NCHW = 3 };

__global__ __launch_bounds__(256) void gemm_bf16_wmma(
    const __bf16* __restrict__ A, const __bf16* __restrict__ Bw,
    int M, int N, int K,
    float* __restrict__ outF, __bf16* __restrict__ outB,
    const float* __restrict__ bias, const float* __restrict__ resid,
    int epi) {
  __shared__ __align__(16) __bf16 As[2][GBM * LDT];
  __shared__ __align__(16) __bf16 Bs[2][GBN * LDT];

  const int tid  = threadIdx.x;
  const int lane = tid & 31;
  const int wave = tid >> 5;
  const int lmod = lane & 15;
  const int half = lane >> 4;                 // 0: lanes 0-15, 1: lanes 16-31

  const int bm = blockIdx.x * GBM;
  const int bn = blockIdx.y * GBN;
  const int wm = (wave >> 2) * 32;            // 0 or 32
  const int wn = (wave & 3) * 32;             // 0,32,64,96

  v8f acc[2][2];
  const v8f vzero = {0.f,0.f,0.f,0.f,0.f,0.f,0.f,0.f};
  #pragma unroll
  for (int i = 0; i < 2; ++i)
    #pragma unroll
    for (int j = 0; j < 2; ++j) acc[i][j] = vzero;

  // N-edge handling (only the proj GEMM, N=48): zero both B buffers once; rows
  // >= N are never written afterwards, so they stay zero for every K step.
  if (bn + GBN > N) {
    uint4 z4 = make_uint4(0u, 0u, 0u, 0u);
    #pragma unroll
    for (int r = 0; r < 6; ++r) {
      int e = (tid + (r << 8)) << 3;          // bf16 element index, 8 per store
      if (e < 2 * GBN * LDT) *(uint4*)&(((__bf16*)Bs)[e]) = z4;
    }
    __syncthreads();
  }

  // Stage one K-tile into LDS buffer `buf`.
  auto stage = [&](int buf, int k0) {
    {
      int row = tid >> 2;                     // 0..63
      int q   = (tid & 3) << 3;               // 0,8,16,24 (bf16 elems)
      const __bf16* src = A + (size_t)(bm + row) * K + k0 + q;
#if USE_ASYNC_LDS
      __builtin_amdgcn_global_load_async_to_lds_b128(
          (gptr_v4i)(void*)src,
          (lptr_v4i)(void*)&As[buf][row * LDT + q], 0, 0);
#else
      *(uint4*)&As[buf][row * LDT + q] = *(const uint4*)src;
#endif
    }
    #pragma unroll
    for (int r = 0; r < 2; ++r) {
      int idx = tid + (r << 8);               // 0..511
      int row = idx >> 2;                     // 0..127
      int q   = (idx & 3) << 3;
      int n   = bn + row;
#if USE_ASYNC_LDS
      if (n < N) {
        const __bf16* src = Bw + (size_t)n * K + k0 + q;
        __builtin_amdgcn_global_load_async_to_lds_b128(
            (gptr_v4i)(void*)src,
            (lptr_v4i)(void*)&Bs[buf][row * LDT + q], 0, 0);
      }
#else
      uint4 val = make_uint4(0u, 0u, 0u, 0u);
      if (n < N) val = *(const uint4*)(Bw + (size_t)n * K + k0 + q);
      *(uint4*)&Bs[buf][row * LDT + q] = val;
#endif
    }
  };

  // WMMA on LDS buffer `buf` (fragment layouts per CDNA5 ISA 16-bit A/B maps).
  auto compute = [&](int buf) {
    FragBF afr[2], bfr[2];
    const int akb = half ? 8 : 0;
    #pragma unroll
    for (int mi = 0; mi < 2; ++mi) {
      int row = wm + mi * 16 + lmod;
      #pragma unroll
      for (int i = 0; i < 8; ++i) {
        int kk = akb + ((i < 4) ? (2 * i) : (16 + 2 * (i - 4)));
        afr[mi].u[i] = *(const unsigned*)&As[buf][row * LDT + kk];
      }
    }
    const int bkb = half ? 8 : 0;
    #pragma unroll
    for (int ni = 0; ni < 2; ++ni) {
      int col = wn + ni * 16 + lmod;
      #pragma unroll
      for (int i = 0; i < 8; ++i)
        bfr[ni].u[i] = *(const unsigned*)&Bs[buf][col * LDT + bkb + ((i < 4) ? (2 * i) : (16 + 2 * (i - 4)))];
    }
    #pragma unroll
    for (int mi = 0; mi < 2; ++mi)
      #pragma unroll
      for (int ni = 0; ni < 2; ++ni)
        acc[mi][ni] = __builtin_amdgcn_wmma_f32_16x16x32_bf16(
            false, afr[mi].v, false, bfr[ni].v, (short)0, acc[mi][ni], false, false);
  };

#if USE_ASYNC_LDS
  // Two-stage async pipeline: global->LDS DMA for tile k+1 overlaps WMMA on tile k.
  stage(0, 0);
  wait_asynccnt0();
  __syncthreads();
  int cur = 0;
  for (int k0 = 0; k0 < K; k0 += GBK) {
    if (k0 + GBK < K) stage(cur ^ 1, k0 + GBK);
    compute(cur);
    wait_asynccnt0();
    __syncthreads();
    cur ^= 1;
  }
#else
  for (int k0 = 0; k0 < K; k0 += GBK) {
    stage(0, k0);
    __syncthreads();
    compute(0);
    __syncthreads();
  }
#endif

  // Epilogue. C layout: VGPR r holds M = r (lanes 0-15) / r+8 (lanes 16-31), N = lmod.
  #pragma unroll
  for (int mi = 0; mi < 2; ++mi) {
    #pragma unroll
    for (int ni = 0; ni < 2; ++ni) {
      int col = bn + wn + ni * 16 + lmod;
      if (col >= N) continue;
      #pragma unroll
      for (int r = 0; r < 8; ++r) {
        int row = bm + wm + mi * 16 + r + half * 8;
        float v = acc[mi][ni][r];
        size_t o = (size_t)row * N + col;
        if (epi == EPI_STORE) {
          outF[o] = v;
        } else if (epi == EPI_RESID) {
          v += resid[o];
          outF[o] = v;
          outB[o] = f2bf(v);
        } else if (epi == EPI_BIAS_GELU) {
          v = geluf(v + bias[col]);
          outB[o] = f2bf(v);
        } else { // EPI_BIAS_RESID_NCHW
          v += bias[col] + resid[o];
          int b = row / LTOK;
          int l = row - b * LTOK;
          outF[((size_t)(b * N + col)) * LTOK + l] = v;
        }
      }
    }
  }
}

// ---------------- depthwise causal conv (k=4) + SiLU ----------------

__global__ void conv_kernel(const float* __restrict__ xz, const float* __restrict__ cw,
                            const float* __restrict__ cb, float* __restrict__ uF,
                            __bf16* __restrict__ uB, int total) {
  int idx = blockIdx.x * blockDim.x + threadIdx.x;
  if (idx >= total) return;
  int row = idx >> 9;            // b*LTOK + l
  int d   = idx & (DI - 1);
  int l   = row & (LTOK - 1);
  float acc = cb[d];
  #pragma unroll
  for (int j = 0; j < 4; ++j) {
    int ll = l + j - 3;
    if (ll >= 0) acc += xz[(size_t)(row + j - 3) * (2 * DI) + d] * cw[d * 4 + j];
  }
  float s = siluf(acc);
  uF[idx] = s;
  uB[idx] = f2bf(s);
}

// ---------------- dt = softplus(proj[:, :R] @ W_dt^T + b_dt) ----------------

__global__ void dt_kernel(const float* __restrict__ proj, const float* __restrict__ W_dt,
                          const float* __restrict__ b_dt, float* __restrict__ dt, int total) {
  int idx = blockIdx.x * blockDim.x + threadIdx.x;
  if (idx >= total) return;
  int row = idx >> 9;
  int d   = idx & (DI - 1);
  const float* p = proj + (size_t)row * PROJW;
  float s = b_dt[d];
  #pragma unroll
  for (int r = 0; r < RNK; ++r) s += p[r] * W_dt[d * RNK + r];
  dt[idx] = softplusf(s);
}

// ---------------- selective scan: one lane per (b,d,n) ----------------

__global__ __launch_bounds__(256) void scan_kernel(
    const float* __restrict__ dt, const float* __restrict__ u, const float* __restrict__ proj,
    const float* __restrict__ A_log, float* __restrict__ ys) {
  int gid  = blockIdx.x * 256 + threadIdx.x;   // 0 .. B*DI*NST-1
  int n    = gid & (NST - 1);
  int pair = gid >> 4;                         // b*DI + d
  int d    = pair & (DI - 1);
  int b    = pair >> 9;

  float A = -__expf(A_log[d * NST + n]);
  float h = 0.f;
  size_t base = (size_t)b * LTOK;
  for (int l = 0; l < LTOK; ++l) {
    size_t row = base + l;
    // Speculative prefetch (global_prefetch_b8): drop-on-fault safe past the end.
    __builtin_prefetch(&dt[(row + 32) * DI + d], 0, 3);
    __builtin_prefetch(&proj[(row + 32) * PROJW + RNK + n], 0, 3);
    float dtv = dt[row * DI + d];
    float uv  = u[row * DI + d];
    float Bv  = proj[row * PROJW + RNK + n];
    float Cv  = proj[row * PROJW + RNK + NST + n];
    float dA  = __expf(dtv * A);
    h = dA * h + (dtv * uv) * Bv;
    float y = h * Cv;
    // reduce over the 16 n-lanes sharing this (b,d)
    y += __shfl_xor(y, 1, 32);
    y += __shfl_xor(y, 2, 32);
    y += __shfl_xor(y, 4, 32);
    y += __shfl_xor(y, 8, 32);
    if (n == 0) ys[row * DI + d] = y;
  }
}

// ---------------- ym = (ys + u*D) * silu(z) ----------------

__global__ void combine_kernel(const float* __restrict__ ys, const float* __restrict__ u,
                               const float* __restrict__ Dvec, const float* __restrict__ xz,
                               __bf16* __restrict__ ymB, int total) {
  int idx = blockIdx.x * blockDim.x + threadIdx.x;
  if (idx >= total) return;
  int row = idx >> 9;
  int d   = idx & (DI - 1);
  float z = xz[(size_t)row * (2 * DI) + DI + d];
  float v = (ys[idx] + u[idx] * Dvec[d]) * siluf(z);
  ymB[idx] = f2bf(v);
}

// ---------------- host launcher ----------------

extern "C" void kernel_launch(void* const* d_in, const int* in_sizes, int n_in,
                              void* d_out, int out_size, void* d_ws, size_t ws_size,
                              hipStream_t stream) {
  const float* x      = (const float*)d_in[0];
  const float* gamma  = (const float*)d_in[1];
  const float* beta   = (const float*)d_in[2];
  const float* W_in   = (const float*)d_in[3];
  const float* conv_w = (const float*)d_in[4];
  const float* conv_b = (const float*)d_in[5];
  const float* W_xp   = (const float*)d_in[6];
  const float* W_dt   = (const float*)d_in[7];
  const float* b_dt   = (const float*)d_in[8];
  const float* A_log  = (const float*)d_in[9];
  const float* Dvec   = (const float*)d_in[10];
  const float* W_out  = (const float*)d_in[11];
  const float* W1     = (const float*)d_in[12];
  const float* b1     = (const float*)d_in[13];
  const float* W2     = (const float*)d_in[14];
  const float* b2     = (const float*)d_in[15];
  float* out = (float*)d_out;

  const int Bb   = in_sizes[0] / (C_DIM * LTOK);   // 2
  const int rows = Bb * LTOK;                      // 8192

  char* ws = (char*)d_ws;
  size_t off = 0;
  auto alloc = [&](size_t bytes) -> char* {
    char* p = ws + off;
    off = (off + bytes + 255) & ~(size_t)255;
    return p;
  };

  float*  ynF   = (float*) alloc((size_t)rows * C_DIM * 4);
  __bf16* ynB   = (__bf16*)alloc((size_t)rows * C_DIM * 2);
  float*  xzF   = (float*) alloc((size_t)rows * 2 * DI * 4);
  float*  uF    = (float*) alloc((size_t)rows * DI * 4);
  __bf16* uB    = (__bf16*)alloc((size_t)rows * DI * 2);
  float*  projF = (float*) alloc((size_t)rows * PROJW * 4);
  float*  dtF   = (float*) alloc((size_t)rows * DI * 4);
  float*  ysF   = (float*) alloc((size_t)rows * DI * 4);
  __bf16* ymB   = (__bf16*)alloc((size_t)rows * DI * 2);
  float*  y2F   = (float*) alloc((size_t)rows * C_DIM * 4);
  __bf16* y2B   = (__bf16*)alloc((size_t)rows * C_DIM * 2);
  __bf16* h1B   = (__bf16*)alloc((size_t)rows * 4 * C_DIM * 2);
  __bf16* WinB  = (__bf16*)alloc((size_t)2 * DI * C_DIM * 2);
  __bf16* WxpB  = (__bf16*)alloc((size_t)PROJW * DI * 2);
  __bf16* WoutB = (__bf16*)alloc((size_t)C_DIM * DI * 2);
  __bf16* W1B   = (__bf16*)alloc((size_t)4 * C_DIM * C_DIM * 2);
  __bf16* W2B   = (__bf16*)alloc((size_t)C_DIM * 4 * C_DIM * 2);
  (void)ws_size; (void)n_in; (void)out_size;

  auto cvt = [&](const float* src, __bf16* dst, int n) {
    cvt_bf16_kernel<<<(n + 255) / 256, 256, 0, stream>>>(src, dst, n);
  };
  cvt(W_in,  WinB,  2 * DI * C_DIM);
  cvt(W_xp,  WxpB,  PROJW * DI);
  cvt(W_out, WoutB, C_DIM * DI);
  cvt(W1,    W1B,   4 * C_DIM * C_DIM);
  cvt(W2,    W2B,   C_DIM * 4 * C_DIM);

  ln_kernel<<<rows, 256, 0, stream>>>(x, gamma, beta, ynF, ynB);

  auto gemm = [&](const __bf16* A, const __bf16* Bw, int M, int N, int K,
                  float* oF, __bf16* oB, const float* bias, const float* resid, int epi) {
    dim3 g(M / GBM, (N + GBN - 1) / GBN);
    gemm_bf16_wmma<<<g, 256, 0, stream>>>(A, Bw, M, N, K, oF, oB, bias, resid, epi);
  };

  // xz = yn @ W_in^T  [8192 x 1024 x 256]
  gemm(ynB, WinB, rows, 2 * DI, C_DIM, xzF, nullptr, nullptr, nullptr, EPI_STORE);
  // depthwise causal conv + SiLU
  conv_kernel<<<(rows * DI) / 256, 256, 0, stream>>>(xzF, conv_w, conv_b, uF, uB, rows * DI);
  // proj = u @ W_xp^T  [8192 x 48 x 512]
  gemm(uB, WxpB, rows, PROJW, DI, projF, nullptr, nullptr, nullptr, EPI_STORE);
  // dt = softplus(proj[:, :R] @ W_dt^T + b_dt)
  dt_kernel<<<(rows * DI) / 256, 256, 0, stream>>>(projF, W_dt, b_dt, dtF, rows * DI);
  // selective scan
  scan_kernel<<<(Bb * DI * NST) / 256, 256, 0, stream>>>(dtF, uF, projF, A_log, ysF);
  // ym = (ys + u*D) * silu(z)
  combine_kernel<<<(rows * DI) / 256, 256, 0, stream>>>(ysF, uF, Dvec, xzF, ymB, rows * DI);
  // y2 = ym @ W_out^T + yn   [8192 x 256 x 512]
  gemm(ymB, WoutB, rows, C_DIM, DI, y2F, y2B, nullptr, ynF, EPI_RESID);
  // h1 = gelu(y2 @ W1^T + b1)  [8192 x 1024 x 256]
  gemm(y2B, W1B, rows, 4 * C_DIM, C_DIM, nullptr, h1B, b1, nullptr, EPI_BIAS_GELU);
  // y3 = h1 @ W2^T + b2 + y2 -> NCHW output  [8192 x 256 x 1024]
  gemm(h1B, W2B, rows, C_DIM, 4 * C_DIM, out, nullptr, b2, y2F, EPI_BIAS_RESID_NCHW);
}
